// GraphConvolution_19782619365995
// MI455X (gfx1250) — matrix operand order repaired and checked by reading, not compiled, not run on hardware
//
#include <hip/hip_runtime.h>
#include <hip/hip_bf16.h>
#include <stdint.h>

// Problem constants (fixed by the reference).
#define N_NODES   50000
#define N_EDGES   1600000
#define DIM       256
#define ROW_TILES 3125        // N_NODES / 16 (exact)

typedef __attribute__((ext_vector_type(16))) __bf16 v16bf;
typedef __attribute__((ext_vector_type(8)))  float  v8f;

#if defined(__has_builtin)
#if __has_builtin(__builtin_amdgcn_cvt_pk_bf16_f32)
#define HAVE_CVT_PK_BF16 1
#endif
#endif

__device__ __forceinline__ unsigned bf16_rne(float f) {
  unsigned u = __float_as_uint(f);
  return (u + 0x7FFFu + ((u >> 16) & 1u)) >> 16;   // round-to-nearest-even bf16
}
__device__ __forceinline__ unsigned pack_bf16x2(float lo, float hi) {
#if HAVE_CVT_PK_BF16
  auto p = __builtin_amdgcn_cvt_pk_bf16_f32(lo, hi);   // single v_cvt_pk_bf16_f32
  return __builtin_bit_cast(unsigned, p);
#else
  return bf16_rne(lo) | (bf16_rne(hi) << 16);
#endif
}
__device__ __forceinline__ float bf_lo(unsigned u) { return __uint_as_float(u << 16); }
__device__ __forceinline__ float bf_hi(unsigned u) { return __uint_as_float(u & 0xFFFF0000u); }

union FragU { unsigned u[8]; v16bf v; };

// ------------------------------------------------------------------
// Kernel 1: support[N,256] (bf16) = x[N,256] @ W[256,256]^T via WMMA.
// Block = 256 threads = 8 waves; each wave owns a 16-row strip and all
// 16 column tiles (16 v8f accumulators). W k-slices staged in LDS in
// B-fragment layout [ntile][v][lane] (bank-conflict-free ds_load_b32).
// ------------------------------------------------------------------
__global__ __launch_bounds__(256) void gemm_xWt(
    const float* __restrict__ x, const float* __restrict__ W,
    unsigned short* __restrict__ support)
{
  __shared__ unsigned sB[16 * 8 * 32];   // 16KB: [ntile][v][lane] packed bf16x2

  const int tid  = threadIdx.x;
  const int wave = tid >> 5;
  const int lane = tid & 31;
  const int m    = lane & 15;
  const int half = lane >> 4;

  const int  rowTile = blockIdx.x * 8 + wave;
  const bool active  = rowTile < ROW_TILES;
  const int  row0    = (active ? rowTile : 0) * 16;   // clamp keeps loads in-bounds
  const size_t arow  = (size_t)(row0 + m) * DIM;

  v8f acc[16];
#pragma unroll
  for (int t = 0; t < 16; ++t)
    acc[t] = (v8f){0.f, 0.f, 0.f, 0.f, 0.f, 0.f, 0.f, 0.f};

  for (int kc = 0; kc < DIM; kc += 32) {
    __syncthreads();   // protect previous iteration's sB reads
    // Stage W[:, kc:kc+32] into LDS as bf16 B-fragments.
    // B k order chosen here (16-bit, 32x16, contiguous, not AMD's table order): lanes 0-15 -> K=0..15, lanes 16-31 -> K=16..31;
    // VGPR v holds K = 2v, 2v+1 within the half. One float4 covers v, v+1.
#pragma unroll
    for (int it = 0; it < 8; ++it) {
      int idx = tid + it * 256;         // 0..2047 (pairs of fragment dwords)
      int nt  = idx >> 7;               // /128
      int v2  = (idx >> 5) & 3;         // v = 2*v2
      int l   = idx & 31;
      int o   = nt * 16 + (l & 15);
      int k   = kc + 4 * v2 + 16 * (l >> 4);
      const float4 w4 = *(const float4*)(W + o * DIM + k);
      sB[(nt * 8 + 2 * v2 + 0) * 32 + l] = pack_bf16x2(w4.x, w4.y);
      sB[(nt * 8 + 2 * v2 + 1) * 32 + l] = pack_bf16x2(w4.z, w4.w);
    }
    __syncthreads();

    // A fragment as packed here (contiguous k order, 16x32): VGPR v, half h -> K = 2v + 16h.
    // v=0..7 cover 16 consecutive K at kc+16h (matches B staging).
    FragU A;
    {
      const float4 q0 = *(const float4*)(x + arow + kc + 16 * half + 0);
      const float4 q1 = *(const float4*)(x + arow + kc + 16 * half + 4);
      const float4 q2 = *(const float4*)(x + arow + kc + 16 * half + 8);
      const float4 q3 = *(const float4*)(x + arow + kc + 16 * half + 12);
      A.u[0] = pack_bf16x2(q0.x, q0.y);  A.u[1] = pack_bf16x2(q0.z, q0.w);
      A.u[2] = pack_bf16x2(q1.x, q1.y);  A.u[3] = pack_bf16x2(q1.z, q1.w);
      A.u[4] = pack_bf16x2(q2.x, q2.y);  A.u[5] = pack_bf16x2(q2.z, q2.w);
      A.u[6] = pack_bf16x2(q3.x, q3.y);  A.u[7] = pack_bf16x2(q3.z, q3.w);
    }

#pragma unroll
    for (int nt = 0; nt < 16; ++nt) {
      FragU B;
#pragma unroll
      for (int v = 0; v < 8; ++v) B.u[v] = sB[(nt * 8 + v) * 32 + lane];
      acc[nt] = __builtin_amdgcn_wmma_f32_16x16x32_bf16(
          false, A.v, false, B.v, (short)0, acc[nt], false, false);
    }
  }

  if (active) {
    // D layout: lanes 0-15: M=r, N=lane; lanes 16-31: M=r+8, N=lane-16.
#pragma unroll
    for (int nt = 0; nt < 16; ++nt) {
#pragma unroll
      for (int r = 0; r < 8; ++r) {
        int mm = r + 8 * half;
        int nn = nt * 16 + m;
        support[(size_t)(row0 + mm) * DIM + nn] = (unsigned short)bf16_rne(acc[nt][r]);
      }
    }
  }
}

// ------------------------------------------------------------------
// CSR construction: zero -> histogram -> exclusive scan -> slot fill.
// ------------------------------------------------------------------
__global__ __launch_bounds__(256) void zero2_kernel(int* __restrict__ a,
                                                    int* __restrict__ b, int n) {
  int i = blockIdx.x * 256 + threadIdx.x;
  if (i < n) { a[i] = 0; b[i] = 0; }
}

__global__ __launch_bounds__(256) void hist_kernel(const int* __restrict__ edge_row,
                                                   int* __restrict__ counts) {
  int e = blockIdx.x * 256 + threadIdx.x;     // E == 6250*256 exactly
  atomicAdd(&counts[edge_row[e]], 1);
}

__global__ __launch_bounds__(1024) void scan_kernel(const int* __restrict__ counts,
                                                    int* __restrict__ row_ptr) {
  __shared__ int sm[1024];
  __shared__ int carry;
  const int tid = threadIdx.x;
  if (tid == 0) carry = 0;
  __syncthreads();
  for (int base = 0; base < N_NODES; base += 1024) {
    int idx = base + tid;
    int v = (idx < N_NODES) ? counts[idx] : 0;
    sm[tid] = v;
    __syncthreads();
    for (int off = 1; off < 1024; off <<= 1) {
      int t = (tid >= off) ? sm[tid - off] : 0;
      __syncthreads();
      sm[tid] += t;
      __syncthreads();
    }
    int incl = sm[tid];
    if (idx < N_NODES) row_ptr[idx] = carry + incl - v;   // exclusive prefix
    __syncthreads();
    if (tid == 1023) carry += incl;                       // chunk total
    __syncthreads();
  }
  if (tid == 0) row_ptr[N_NODES] = carry;                 // == N_EDGES
}

__global__ __launch_bounds__(256) void fill_kernel(
    const int* __restrict__ edge_row, const int* __restrict__ row_ptr,
    int* __restrict__ fillc, int* __restrict__ csr_e) {
  int e = blockIdx.x * 256 + threadIdx.x;
  int r = edge_row[e];
  int s = atomicAdd(&fillc[r], 1);
  csr_e[row_ptr[r] + s] = e;
}

// ------------------------------------------------------------------
// Kernel 5: out[row] = sum_{e in row} val[e] * support[col[e]].
// One wave per destination row; lane owns 8 contiguous floats in
// registers. b128 bf16 gathers (L2-resident), zero atomics, b128 stores.
// ------------------------------------------------------------------
__global__ __launch_bounds__(256) void gather_kernel(
    const unsigned short* __restrict__ support,
    const int* __restrict__ edge_col, const float* __restrict__ edge_val,
    const int* __restrict__ row_ptr, const int* __restrict__ csr_e,
    float* __restrict__ out)
{
  const int wave = threadIdx.x >> 5;
  const int lane = threadIdx.x & 31;
  const int row  = blockIdx.x * 8 + wave;     // 6250*8 == N_NODES exactly
  const int beg  = row_ptr[row];
  const int end  = row_ptr[row + 1];
  const int coff = lane * 8;

  float a0 = 0.f, a1 = 0.f, a2 = 0.f, a3 = 0.f;
  float a4 = 0.f, a5 = 0.f, a6 = 0.f, a7 = 0.f;

  for (int i = beg; i < end; ++i) {
    const int   e = csr_e[i];
    const int   c = edge_col[e];
    const float v = edge_val[e];
    const uint4 d = *(const uint4*)(support + (size_t)c * DIM + coff);
    a0 = fmaf(v, bf_lo(d.x), a0);  a1 = fmaf(v, bf_hi(d.x), a1);
    a2 = fmaf(v, bf_lo(d.y), a2);  a3 = fmaf(v, bf_hi(d.y), a3);
    a4 = fmaf(v, bf_lo(d.z), a4);  a5 = fmaf(v, bf_hi(d.z), a5);
    a6 = fmaf(v, bf_lo(d.w), a6);  a7 = fmaf(v, bf_hi(d.w), a7);
  }

  float4* op = (float4*)(out + (size_t)row * DIM + coff);
  op[0] = (float4){a0, a1, a2, a3};
  op[1] = (float4){a4, a5, a6, a7};
}

// ------------------------------------------------------------------
// Workspace layout (bytes):
//   [0, 25600000)            support bf16  (50000*256*2)
//   [25600000, 25800000)     counts        (50000*4)
//   [25800000, 26000004)     row_ptr       (50001*4)
//   [26000128, 26200128)     fillc         (50000*4)
//   [26200128, 32600128)     csr_e         (1600000*4)
// ------------------------------------------------------------------
extern "C" void kernel_launch(void* const* d_in, const int* in_sizes, int n_in,
                              void* d_out, int out_size, void* d_ws, size_t ws_size,
                              hipStream_t stream)
{
  const float* x        = (const float*)d_in[0];
  const float* W        = (const float*)d_in[1];
  const int*   edge_row = (const int*)d_in[2];
  const int*   edge_col = (const int*)d_in[3];
  const float* edge_val = (const float*)d_in[4];
  float* out = (float*)d_out;

  char* ws = (char*)d_ws;
  unsigned short* support = (unsigned short*)(ws);
  int* counts  = (int*)(ws + 25600000);
  int* row_ptr = (int*)(ws + 25800000);
  int* fillc   = (int*)(ws + 26000128);
  int* csr_e   = (int*)(ws + 26200128);

  zero2_kernel <<<196, 256, 0, stream>>>(counts, fillc, N_NODES);
  gemm_xWt     <<<391, 256, 0, stream>>>(x, W, support);
  hist_kernel  <<<N_EDGES / 256, 256, 0, stream>>>(edge_row, counts);
  scan_kernel  <<<1, 1024, 0, stream>>>(counts, row_ptr);
  fill_kernel  <<<N_EDGES / 256, 256, 0, stream>>>(edge_row, row_ptr, fillc, csr_e);
  gather_kernel<<<N_NODES / 8, 256, 0, stream>>>(support, edge_col, edge_val,
                                                 row_ptr, csr_e, out);
}
